// Delphi_48249662603386
// MI455X (gfx1250) — matrix operand order repaired and checked by reading, not compiled, not run on hardware
//
#include <hip/hip_runtime.h>
#include <hip/hip_bf16.h>
#include <math.h>

#define LL long long

// Model dims (fixed by the reference)
constexpr int Lr = 12;          // layers
constexpr int Hh = 16;          // heads
constexpr int Cc = 1024;        // channels
constexpr int Tt = 512;         // sequence
constexpr int Bb = 4;           // batch
constexpr int Vv = 1280;        // vocab
constexpr int BT = Bb * Tt;     // 2048 token rows
constexpr int HD = Cc / Hh;     // 64 head dim

typedef __attribute__((ext_vector_type(16))) _Float16 v16h;
typedef __attribute__((ext_vector_type(8)))  _Float16 h8;
typedef __attribute__((ext_vector_type(8)))  float    v8f;
typedef __attribute__((ext_vector_type(4)))  int      i4;

// ---- CDNA5 async global->LDS path (graceful fallback) ---------------------
#if defined(__has_builtin)
#if __has_builtin(__builtin_amdgcn_global_load_async_to_lds_b128)
#define HAVE_ASYNC_LDS 1
#endif
#endif
#ifndef HAVE_ASYNC_LDS
#define HAVE_ASYNC_LDS 0
#endif

#if HAVE_ASYNC_LDS
// copy one 16-byte chunk global -> LDS via GLOBAL_LOAD_ASYNC_TO_LDS_B128
#define ASYNC_CP16(gsrc, ldst)                                                 \
  __builtin_amdgcn_global_load_async_to_lds_b128(                              \
      (__attribute__((address_space(1))) i4*)(uintptr_t)(gsrc),                \
      (__attribute__((address_space(3))) i4*)(uintptr_t)(ldst), 0, 0)
__device__ __forceinline__ void async_wait0() {
#if __has_builtin(__builtin_amdgcn_s_wait_asynccnt)
  __builtin_amdgcn_s_wait_asynccnt(0);
#endif
}
#endif

// ---------------------------------------------------------------------------
// fp32 -> f16 weight conversion (grid-stride)
// ---------------------------------------------------------------------------
__global__ __launch_bounds__(256)
void k_cvt_f16(const float* __restrict__ in, _Float16* __restrict__ out, LL n) {
  LL i = (LL)blockIdx.x * blockDim.x + threadIdx.x;
  LL stride = (LL)gridDim.x * blockDim.x;
  for (; i < n; i += stride) out[i] = (_Float16)in[i];
}

// ---------------------------------------------------------------------------
// x = wte[idx] + wme[modality] + sinusoidal_age_encoding   (fp32 residual)
// ---------------------------------------------------------------------------
__global__ __launch_bounds__(256)
void k_embed(const int* __restrict__ idx, const float* __restrict__ age,
             const int* __restrict__ mod, const float* __restrict__ wte,
             const float* __restrict__ wme, float* __restrict__ x) {
  int row = blockIdx.x;                       // [0, B*T)
  int tok = idx[row], m = mod[row];
  float a = age[row];
  const int half = Cc / 2;
  const float lw = 9.210340371976184f / (float)half;   // ln(10000)/half
  for (int c = threadIdx.x; c < Cc; c += blockDim.x) {
    int f = (c < half) ? c : (c - half);
    float ang = a * __expf(-(float)f * lw);
    float pe = (c < half) ? __sinf(ang) : __cosf(ang);
    x[(LL)row * Cc + c] = wte[(LL)tok * Cc + c] + wme[(LL)m * Cc + c] + pe;
  }
}

// ---------------------------------------------------------------------------
// row layernorm over C=1024 (fp32 in, f16 out for WMMA consumption)
// ---------------------------------------------------------------------------
__global__ __launch_bounds__(256)
void k_layernorm(const float* __restrict__ x, const float* __restrict__ w,
                 const float* __restrict__ b, _Float16* __restrict__ out) {
  __shared__ float red[256];
  int row = blockIdx.x, tid = threadIdx.x;
  const float* xr = x + (LL)row * Cc;
  float s = 0.f;
  for (int c = tid; c < Cc; c += 256) s += xr[c];
  red[tid] = s; __syncthreads();
  for (int o = 128; o > 0; o >>= 1) { if (tid < o) red[tid] += red[tid + o]; __syncthreads(); }
  float mu = red[0] / (float)Cc; __syncthreads();
  float v = 0.f;
  for (int c = tid; c < Cc; c += 256) { float d = xr[c] - mu; v += d * d; }
  red[tid] = v; __syncthreads();
  for (int o = 128; o > 0; o >>= 1) { if (tid < o) red[tid] += red[tid + o]; __syncthreads(); }
  float rs = rsqrtf(red[0] / (float)Cc + 1e-5f);
  for (int c = tid; c < Cc; c += 256)
    out[(LL)row * Cc + c] = (_Float16)((xr[c] - mu) * rs * w[c] + b[c]);
}

// ---------------------------------------------------------------------------
// in-place masked softmax on a [T] row of the atts slab in d_out.
// atts slab >> L2 (67MB/layer), so use non-temporal accesses.
// mask: causal AND (ages differ OR diagonal)
// ---------------------------------------------------------------------------
__global__ __launch_bounds__(256)
void k_softmax_mask(float* __restrict__ att, const float* __restrict__ age) {
  __shared__ float red[256];
  LL rowid = blockIdx.x;                      // b*H*T + h*T + q
  int q = (int)(rowid % Tt);
  int b = (int)(rowid / ((LL)Tt * Hh));
  float* row = att + rowid * Tt;
  float aq = age[(LL)b * Tt + q];
  int tid = threadIdx.x;

  int k0 = tid, k1 = tid + 256;
  bool ok0 = (k0 <= q) && ((age[(LL)b * Tt + k0] != aq) || (k0 == q));
  bool ok1 = (k1 <= q) && ((age[(LL)b * Tt + k1] != aq) || (k1 == q));
  float v0 = ok0 ? __builtin_nontemporal_load(&row[k0]) : -INFINITY;
  float v1 = ok1 ? __builtin_nontemporal_load(&row[k1]) : -INFINITY;

  red[tid] = fmaxf(v0, v1); __syncthreads();
  for (int o = 128; o > 0; o >>= 1) { if (tid < o) red[tid] = fmaxf(red[tid], red[tid + o]); __syncthreads(); }
  float mx = red[0]; __syncthreads();

  float e0 = __expf(v0 - mx), e1 = __expf(v1 - mx);   // -inf -> 0
  red[tid] = e0 + e1; __syncthreads();
  for (int o = 128; o > 0; o >>= 1) { if (tid < o) red[tid] += red[tid + o]; __syncthreads(); }
  float inv = 1.f / red[0];
  __builtin_nontemporal_store(e0 * inv, &row[k0]);
  __builtin_nontemporal_store(e1 * inv, &row[k1]);
}

__device__ __forceinline__ float gelu_tanh(float v) {
  return 0.5f * v * (1.0f + tanhf(0.7978845608028654f * (v + 0.044715f * v * v * v)));
}

// ---------------------------------------------------------------------------
// Generic WMMA f16 GEMM:  C[m,n] op= alpha * sum_k A[m,k] * W[n,k] (+ bias[n])
//   - 128 threads = 4 wave32; 128x64 output tile; each wave owns 32 rows:
//     2 A-frags x 4 B-frags = 8 v_wmma per K-step (B frags reused twice)
//   - DOUBLE-BUFFERED LDS: async global->LDS loads for K-step k+1 are issued
//     right after the barrier and stay in flight across the WMMA compute of
//     step k; only waited (s_wait_asynccnt 0) at the top of the next step.
//   - AK:  1 = A f16 (unit k-stride, async staged) | 0 = A fp32 (NT, converted)
//   - BS:  0 = B f16 unit k-stride (async staged)  | 1 = B f16 strided k (V)
//   - CMODE: 0 f32 store | 1 f32 non-temporal store | 2 f32 residual +=
//            3 f16 store | 4 f16 gelu store
//   - batched over blockIdx.z = b*Hdim + h with per-operand (batch,head) strides
// All shapes satisfy M%128==0, N%64==0, K%32==0 -> no edge code.
// ---------------------------------------------------------------------------
template<int AK, int BS, int CMODE>
__global__ __launch_bounds__(128)
void k_gemm_wmma(const void* __restrict__ Ap, const _Float16* __restrict__ Bp,
                 const float* __restrict__ bias, void* __restrict__ Cp,
                 int M, int N, int K, float alpha,
                 LL asM, LL aBB, LL aBH,
                 LL bsN, LL bsK, LL bBB, LL bBH,
                 LL csM, LL cBB, LL cBH, int Hdim) {
  __shared__ alignas(16) _Float16 As[2][128][40];  // ping-pong; +8 pad per row
  __shared__ alignas(16) _Float16 Bs[2][64][40];

  const int tid = threadIdx.x;
  const int bbi = blockIdx.z / Hdim, hhi = blockIdx.z % Hdim;
  const int m0 = blockIdx.x * 128, n0 = blockIdx.y * 64;

  const _Float16* Ah = (const _Float16*)Ap + (LL)bbi * aBB + (LL)hhi * aBH;
  const float*    Af = (const float*)Ap    + (LL)bbi * aBB + (LL)hhi * aBH;
  const _Float16* Bh = Bp + (LL)bbi * bBB + (LL)hhi * bBH;

  const int lane = tid & 31, w = tid >> 5;
  const int lr = lane & 15, lhi = lane >> 4;
  const int brow = tid & 63, bhalf = tid >> 6;   // B staging: 2 threads per row

  // stage one 128x32 A tile + 64x32 B tile into LDS buffer `buf`
  auto stage = [&](int buf, int k0) {
    if (AK == 1) {
      const h8* src = (const h8*)(Ah + (LL)(m0 + tid) * asM + k0);
      h8* dst = (h8*)&As[buf][tid][0];
#if HAVE_ASYNC_LDS
      ASYNC_CP16(src + 0, dst + 0); ASYNC_CP16(src + 1, dst + 1);
      ASYNC_CP16(src + 2, dst + 2); ASYNC_CP16(src + 3, dst + 3);
#else
      dst[0] = src[0]; dst[1] = src[1]; dst[2] = src[2]; dst[3] = src[3];
#endif
      __builtin_prefetch(src + 8, 0, 1);         // 2 K-steps ahead (speculative)
    } else {
      const float* srow = Af + (LL)(m0 + tid) * asM + k0;
#pragma unroll
      for (int c = 0; c < 32; ++c)
        As[buf][tid][c] = (_Float16)__builtin_nontemporal_load(&srow[c]);
    }
    if (BS == 0) {
      const h8* src = (const h8*)(Bh + (LL)(n0 + brow) * bsN + k0 + bhalf * 16);
      h8* dst = (h8*)&Bs[buf][brow][bhalf * 16];
#if HAVE_ASYNC_LDS
      ASYNC_CP16(src + 0, dst + 0); ASYNC_CP16(src + 1, dst + 1);
#else
      dst[0] = src[0]; dst[1] = src[1];
#endif
      __builtin_prefetch(src + 8, 0, 1);
    } else {
#pragma unroll
      for (int c = bhalf * 16; c < bhalf * 16 + 16; ++c)
        Bs[buf][brow][c] = Bh[(LL)(n0 + brow) * bsN + (LL)(k0 + c) * bsK];
    }
  };

  v8f acc[2][4] = {};

  stage(0, 0);                                   // preload first tile
  for (int k0 = 0; k0 < K; k0 += 32) {
    const int cur = (k0 >> 5) & 1;
#if HAVE_ASYNC_LDS
    if (AK == 1 || BS == 0) async_wait0();       // tile `cur` resident
#endif
    __syncthreads();
    if (k0 + 32 < K) stage(cur ^ 1, k0 + 32);    // async loads overlap compute

    // ---- fragments (ISA 7.12.2 lane layouts) + 8 WMMAs ----
    v16h af[2];
#pragma unroll
    for (int f = 0; f < 2; ++f) {
      h8 alo = *(const h8*)&As[cur][w * 32 + f * 16 + lr][lhi * 8];
      h8 ahi = *(const h8*)&As[cur][w * 32 + f * 16 + lr][lhi * 8 + 16];
#pragma unroll
      for (int j = 0; j < 8; ++j) { af[f][j] = alo[j]; af[f][j + 8] = ahi[j]; }
    }
#pragma unroll
    for (int nt = 0; nt < 4; ++nt) {
      v16h bf;
      h8 blo = *(const h8*)&Bs[cur][nt * 16 + lr][lhi * 8];
      h8 bhi = *(const h8*)&Bs[cur][nt * 16 + lr][lhi * 8 + 16];
#pragma unroll
      for (int j = 0; j < 8; ++j) { bf[j] = blo[j]; bf[j + 8] = bhi[j]; }
      acc[0][nt] = __builtin_amdgcn_wmma_f32_16x16x32_f16(
          false, af[0], false, bf, (short)0, acc[0][nt], false, false);
      acc[1][nt] = __builtin_amdgcn_wmma_f32_16x16x32_f16(
          false, af[1], false, bf, (short)0, acc[1][nt], false, false);
    }
    __syncthreads();                             // done reading buf `cur`
  }

  // ---- epilogue: VGPR r -> M = r (+8 for lanes 16-31); N = lane&15 ----
  float*    Cf = (float*)Cp    + (LL)bbi * cBB + (LL)hhi * cBH;
  _Float16* Ch = (_Float16*)Cp + (LL)bbi * cBB + (LL)hhi * cBH;
#pragma unroll
  for (int f = 0; f < 2; ++f) {
#pragma unroll
    for (int nt = 0; nt < 4; ++nt) {
      int n = n0 + nt * 16 + lr;
      float bv = bias ? bias[n] : 0.f;
#pragma unroll
      for (int r = 0; r < 8; ++r) {
        int m = m0 + w * 32 + f * 16 + lhi * 8 + r;
        float v = alpha * acc[f][nt][r] + bv;
        LL off = (LL)m * csM + n;
        if (CMODE == 0) Cf[off] = v;
        if (CMODE == 1) __builtin_nontemporal_store(v, &Cf[off]);
        if (CMODE == 2) Cf[off] += v;
        if (CMODE == 3) Ch[off] = (_Float16)v;
        if (CMODE == 4) Ch[off] = (_Float16)gelu_tanh(v);
      }
    }
  }
}

// ---------------------------------------------------------------------------
extern "C" void kernel_launch(void* const* d_in, const int* in_sizes, int n_in,
                              void* d_out, int out_size, void* d_ws, size_t ws_size,
                              hipStream_t stream) {
  (void)in_sizes; (void)n_in; (void)out_size; (void)ws_size;
  const int*   idx   = (const int*)d_in[0];
  const float* age   = (const float*)d_in[1];
  const int*   mod   = (const int*)d_in[2];
  const float* wte   = (const float*)d_in[3];
  const float* wme   = (const float*)d_in[4];
  const float* ln1w  = (const float*)d_in[5];
  const float* ln1b  = (const float*)d_in[6];
  const float* attnw = (const float*)d_in[7];
  const float* attnb = (const float*)d_in[8];
  const float* projw = (const float*)d_in[9];
  const float* projb = (const float*)d_in[10];
  const float* ln2w  = (const float*)d_in[11];
  const float* ln2b  = (const float*)d_in[12];
  const float* fcw   = (const float*)d_in[13];
  const float* fcb   = (const float*)d_in[14];
  const float* fcpw  = (const float*)d_in[15];
  const float* fcpb  = (const float*)d_in[16];
  const float* lnfw  = (const float*)d_in[17];
  const float* lnfb  = (const float*)d_in[18];

  float* out = (float*)d_out;                 // [BT*V logits][L*B*H*T*T atts]

  // ---- workspace carve-up -------------------------------------------------
  char* p = (char*)d_ws;
  auto allocF = [&](LL n) { float* r = (float*)p; p += n * sizeof(float); return r; };
  auto allocH = [&](LL n) { _Float16* r = (_Float16*)p; p += n * sizeof(_Float16); return r; };
  float*    x     = allocF((LL)BT * Cc);            // fp32 residual stream
  _Float16* h16   = allocH((LL)BT * Cc);            // LN outputs
  _Float16* qkv16 = allocH((LL)BT * 3 * Cc);        // fused qkv
  _Float16* y16   = allocH((LL)BT * Cc);            // attention output
  _Float16* mid16 = allocH((LL)BT * 4 * Cc);        // MLP hidden
  _Float16* w_attn = allocH((LL)Lr * 3 * Cc * Cc);
  _Float16* w_proj = allocH((LL)Lr * Cc * Cc);
  _Float16* w_fc   = allocH((LL)Lr * 4 * Cc * Cc);
  _Float16* w_fcp  = allocH((LL)Lr * 4 * Cc * Cc);
  _Float16* w_lm   = allocH((LL)Vv * Cc);

  // ---- per-launch weight down-conversion to f16 ---------------------------
  k_cvt_f16<<<2048, 256, 0, stream>>>(attnw, w_attn, (LL)Lr * 3 * Cc * Cc);
  k_cvt_f16<<<2048, 256, 0, stream>>>(projw, w_proj, (LL)Lr * Cc * Cc);
  k_cvt_f16<<<2048, 256, 0, stream>>>(fcw,   w_fc,   (LL)Lr * 4 * Cc * Cc);
  k_cvt_f16<<<2048, 256, 0, stream>>>(fcpw,  w_fcp,  (LL)Lr * 4 * Cc * Cc);
  k_cvt_f16<<<512,  256, 0, stream>>>(wte,   w_lm,   (LL)Vv * Cc);

  // ---- embedding ----------------------------------------------------------
  k_embed<<<BT, 256, 0, stream>>>(idx, age, mod, wte, wme, x);

  const LL attPerLayer = (LL)Bb * Hh * Tt * Tt;
  const float scale = 0.125f;                 // 1/sqrt(64)

  for (int l = 0; l < Lr; ++l) {
    float* att_l = out + (LL)BT * Vv + (LL)l * attPerLayer;

    // h16 = LN1(x)
    k_layernorm<<<BT, 256, 0, stream>>>(x, ln1w + (LL)l * Cc, ln1b + (LL)l * Cc, h16);

    // qkv16 = h16 @ attn_w^T + attn_b       [2048 x 3072 x 1024] -> f16
    k_gemm_wmma<1, 0, 3><<<dim3(BT / 128, 3 * Cc / 64, 1), 128, 0, stream>>>(
        h16, w_attn + (LL)l * 3 * Cc * Cc, attnb + (LL)l * 3 * Cc, qkv16,
        BT, 3 * Cc, Cc, 1.0f,
        (LL)Cc, 0, 0,
        (LL)Cc, 1, 0, 0,
        (LL)3 * Cc, 0, 0, 1);

    // scores = scale * Q @ K^T -> atts slab (f32, NT)  per (b,h): [512x512x64]
    k_gemm_wmma<1, 0, 1><<<dim3(Tt / 128, Tt / 64, Bb * Hh), 128, 0, stream>>>(
        qkv16 /*Q*/, qkv16 + Cc /*K*/, nullptr, att_l,
        Tt, Tt, HD, scale,
        (LL)3 * Cc, (LL)Tt * 3 * Cc, (LL)HD,
        (LL)3 * Cc, 1, (LL)Tt * 3 * Cc, (LL)HD,
        (LL)Tt, (LL)Hh * Tt * Tt, (LL)Tt * Tt, Hh);

    // masked softmax, in place (this is also the required atts output)
    k_softmax_mask<<<Bb * Hh * Tt, 256, 0, stream>>>(att_l, age);

    // y16 = att @ V   (A fp32 NT, B strided V)   per (b,h): [512 x 64 x 512]
    k_gemm_wmma<0, 1, 3><<<dim3(Tt / 128, 1, Bb * Hh), 128, 0, stream>>>(
        att_l, qkv16 + 2 * Cc /*V*/, nullptr, y16,
        Tt, HD, Tt, 1.0f,
        (LL)Tt, (LL)Hh * Tt * Tt, (LL)Tt * Tt,
        1, (LL)3 * Cc, (LL)Tt * 3 * Cc, (LL)HD,
        (LL)Cc, (LL)Tt * Cc, (LL)HD, Hh);

    // x += y16 @ proj_w^T + proj_b          [2048 x 1024 x 1024]
    k_gemm_wmma<1, 0, 2><<<dim3(BT / 128, Cc / 64, 1), 128, 0, stream>>>(
        y16, w_proj + (LL)l * Cc * Cc, projb + (LL)l * Cc, x,
        BT, Cc, Cc, 1.0f,
        (LL)Cc, 0, 0,
        (LL)Cc, 1, 0, 0,
        (LL)Cc, 0, 0, 1);

    // h16 = LN2(x)
    k_layernorm<<<BT, 256, 0, stream>>>(x, ln2w + (LL)l * Cc, ln2b + (LL)l * Cc, h16);

    // mid16 = gelu(h16 @ fc_w^T + fc_b)     [2048 x 4096 x 1024] -> f16
    k_gemm_wmma<1, 0, 4><<<dim3(BT / 128, 4 * Cc / 64, 1), 128, 0, stream>>>(
        h16, w_fc + (LL)l * 4 * Cc * Cc, fcb + (LL)l * 4 * Cc, mid16,
        BT, 4 * Cc, Cc, 1.0f,
        (LL)Cc, 0, 0,
        (LL)Cc, 1, 0, 0,
        (LL)4 * Cc, 0, 0, 1);

    // x += mid16 @ fcp_w^T + fcp_b          [2048 x 1024 x 4096]
    k_gemm_wmma<1, 0, 2><<<dim3(BT / 128, Cc / 64, 1), 128, 0, stream>>>(
        mid16, w_fcp + (LL)l * Cc * 4 * Cc, fcpb + (LL)l * Cc, x,
        BT, Cc, 4 * Cc, 1.0f,
        (LL)4 * Cc, 0, 0,
        (LL)4 * Cc, 1, 0, 0,
        (LL)Cc, 0, 0, 1);
  }

  // h16 = LN_f(x); logits = h16 @ wte^T     [2048 x 1280 x 1024] -> f32
  k_layernorm<<<BT, 256, 0, stream>>>(x, lnfw, lnfb, h16);
  k_gemm_wmma<1, 0, 0><<<dim3(BT / 128, Vv / 64, 1), 128, 0, stream>>>(
      h16, w_lm, nullptr, out,
      BT, Vv, Cc, 1.0f,
      (LL)Cc, 0, 0,
      (LL)Cc, 1, 0, 0,
      (LL)Vv, 0, 0, 1);
}
